// MultiheadAttention_48249662603809
// MI455X (gfx1250) — matrix operand order repaired and checked by reading, not compiled, not run on hardware
//
#include <hip/hip_runtime.h>
#include <hip/hip_bf16.h>

typedef __bf16 bf16_t;
typedef bf16_t v16bf __attribute__((ext_vector_type(16)));
typedef bf16_t v8bf  __attribute__((ext_vector_type(8)));
typedef float  v8f   __attribute__((ext_vector_type(8)));
typedef float  v4f   __attribute__((ext_vector_type(4)));

#define WMMA_BF16(a, b, c) \
  __builtin_amdgcn_wmma_f32_16x16x32_bf16(false, (a), false, (b), (short)0, (c), false, false)

// ---------------------------------------------------------------------------
// Fragment loaders (CDNA5 WMMA VGPR layouts, ISA 7.12.2)
// A: 16x32 bf16 (MxK). lanes 0-15: M=lane, K = {0..7,16..23}; lanes 16-31: same rows, K = {8..15,24..31}
// ---------------------------------------------------------------------------
__device__ __forceinline__ v16bf load_a16x32(const bf16_t* __restrict__ base, int ld,
                                             int row0, int k0, int lane) {
  int r  = row0 + (lane & 15);
  int kb = k0 + ((lane >> 4) << 3);
  const bf16_t* p = base + (size_t)r * ld + kb;
  v8bf lo = *(const v8bf*)(p);        // K = kb .. kb+7
  v8bf hi = *(const v8bf*)(p + 16);   // K = kb+16 .. kb+23
  return __builtin_shufflevector(lo, hi, 0,1,2,3,4,5,6,7,8,9,10,11,12,13,14,15);
}

// B: 32x16 bf16 (KxN) fetched from N-major storage bt[n][k] (leading dim = K stride).
__device__ __forceinline__ v16bf load_b32x16(const bf16_t* __restrict__ bt, int ld,
                                             int k0, int n0, int lane) {
  int n  = n0 + (lane & 15);
  int kb = k0 + ((lane >> 4) << 3);
  const bf16_t* p = bt + (size_t)n * ld + kb;
  v8bf lo = *(const v8bf*)(p);
  v8bf hi = *(const v8bf*)(p + 16);
  return __builtin_shufflevector(lo, hi, 0,1,2,3,4,5,6,7,8,9,10,11,12,13,14,15);
}

// ---------------------------------------------------------------------------
// Elementwise f32 -> bf16
// ---------------------------------------------------------------------------
__global__ void __launch_bounds__(256) cvt_bf16_kernel(const float* __restrict__ src,
                                                       bf16_t* __restrict__ dst, int n) {
  for (int i = blockIdx.x * blockDim.x + threadIdx.x; i < n; i += gridDim.x * blockDim.x)
    dst[i] = (bf16_t)src[i];
}

// f32 [K][N] row-major -> bf16 transposed [N][K] (N-major for WMMA B fragments)
__global__ void __launch_bounds__(256) cvt_t_bf16_kernel(const float* __restrict__ src,
                                                         bf16_t* __restrict__ dst,
                                                         int K, int N) {
  int total = K * N;
  for (int i = blockIdx.x * blockDim.x + threadIdx.x; i < total; i += gridDim.x * blockDim.x) {
    int k = i / N, n = i - k * N;
    dst[(size_t)n * K + k] = (bf16_t)src[i];
  }
}

// ---------------------------------------------------------------------------
// QKV GEMM: [8192 x 1024] (bf16) @ [1024 x 3072] (as N-major bf16) + bias
// 256 threads = 8 waves; block tile 128x128; wave tile 64x32 (4Mx2N accums).
// Epilogue scatters into Q[B,H,S,64], K[B,H,S,64], Vt[B,H,64,S] (V transposed).
// ---------------------------------------------------------------------------
__global__ void __launch_bounds__(256) qkv_gemm_kernel(const bf16_t* __restrict__ xbf,
                                                       const bf16_t* __restrict__ wt,
                                                       const float* __restrict__ bias,
                                                       bf16_t* __restrict__ Q,
                                                       bf16_t* __restrict__ Kbuf,
                                                       bf16_t* __restrict__ Vt) {
  const int lane = threadIdx.x & 31;
  const int w    = threadIdx.x >> 5;
  const int tm = blockIdx.x * 128, tn = blockIdx.y * 128;
  const int wm = (w & 1) * 64, wn = (w >> 1) * 32;

  v8f acc[4][2] = {};
#pragma unroll 1
  for (int k0 = 0; k0 < 1024; k0 += 32) {
    __builtin_prefetch(xbf + (size_t)(tm + wm) * 1024 + k0 + 64, 0, 1);
    __builtin_prefetch(wt + (size_t)(tn + wn) * 1024 + k0 + 64, 0, 1);
    v16bf a[4];
#pragma unroll
    for (int mi = 0; mi < 4; ++mi)
      a[mi] = load_a16x32(xbf, 1024, tm + wm + mi * 16, k0, lane);
    v16bf b0 = load_b32x16(wt, 1024, k0, tn + wn + 0, lane);
    v16bf b1 = load_b32x16(wt, 1024, k0, tn + wn + 16, lane);
#pragma unroll
    for (int mi = 0; mi < 4; ++mi) {
      acc[mi][0] = WMMA_BF16(a[mi], b0, acc[mi][0]);
      acc[mi][1] = WMMA_BF16(a[mi], b1, acc[mi][1]);
    }
  }

  const int rb = (lane >> 4) << 3, cl = lane & 15;
#pragma unroll
  for (int mi = 0; mi < 4; ++mi) {
#pragma unroll
    for (int nj = 0; nj < 2; ++nj) {
#pragma unroll
      for (int i = 0; i < 8; ++i) {
        int m = tm + wm + mi * 16 + rb + i;
        int n = tn + wn + nj * 16 + cl;
        float v = acc[mi][nj][i] + bias[n];
        int b = m >> 10, s = m & 1023;
        int h = n / 192, t = n - h * 192;
        int which = t >> 6, d = t & 63;
        bf16_t bv = (bf16_t)v;
        size_t bh = (size_t)(b * 16 + h);
        if (which == 0)      Q[(bh * 1024 + s) * 64 + d]    = bv;
        else if (which == 1) Kbuf[(bh * 1024 + s) * 64 + d] = bv;
        else                 Vt[(bh * 64 + d) * 1024 + s]   = bv;  // transposed V
      }
    }
  }
}

// ---------------------------------------------------------------------------
// Fused attention: one WG = one (b,h) x 64 q-rows.
//  - Q tile (8KB) staged to LDS via CDNA5 async global->LDS (ASYNCcnt path)
//  - Logits (64x1024 f32) in 256KB dynamic LDS (CDNA5: 320KB/WGP)
//  - softmax in LDS; attention written to global exactly once
//  - P @ Vt via WMMA, A fragment converted f32->bf16 once per two WMMAs
// ---------------------------------------------------------------------------
__global__ void __launch_bounds__(256) attn_kernel(const bf16_t* __restrict__ Q,
                                                   const bf16_t* __restrict__ Kbuf,
                                                   const bf16_t* __restrict__ Vt,
                                                   float* __restrict__ attn_out,
                                                   bf16_t* __restrict__ vals) {
  extern __shared__ float P[];       // [64][1024] logits / probabilities
  __shared__ bf16_t Qs[64 * 64];     // staged Q tile (8KB)
  __shared__ float red[4][64];
  __shared__ float rowstat[64];

  const int lane = threadIdx.x & 31;
  const int w    = threadIdx.x >> 5;
  const int bh   = blockIdx.x;       // b*16 + h
  const int qblk = blockIdx.y;       // 64-row q block
  const bf16_t* Qb = Q    + (size_t)bh * 1024 * 64;
  const bf16_t* Kc = Kbuf + (size_t)bh * 1024 * 64;
  const bf16_t* Vb = Vt   + (size_t)bh * 64 * 1024;

  // ---- Phase 0: async-copy Q block (64x64 bf16 = 8KB) into LDS.
  // 256 threads x 2 x B128 = 8192 bytes. Tracked by ASYNCcnt.
  {
    const bf16_t* gsrc = Qb + (size_t)qblk * 64 * 64 + threadIdx.x * 16;
    unsigned loff = (unsigned)(uintptr_t)(&Qs[threadIdx.x * 16]);
    asm volatile("global_load_async_to_lds_b128 %0, %1, off"
                 :: "v"(loff), "v"(gsrc) : "memory");
    asm volatile("global_load_async_to_lds_b128 %0, %1, off offset:16"
                 :: "v"(loff), "v"(gsrc) : "memory");
    asm volatile("s_wait_asynccnt 0x0" ::: "memory");
  }
  __syncthreads();

  // ---- Phase 1: logits = scale * Q Kt. Wave w owns key columns [w*128, w*128+128)
  v16bf qa[4][2];
#pragma unroll
  for (int mi = 0; mi < 4; ++mi)
#pragma unroll
    for (int ki = 0; ki < 2; ++ki) {
      int r  = mi * 16 + (lane & 15);
      int kb = ki * 32 + ((lane >> 4) << 3);
      const bf16_t* p = &Qs[r * 64 + kb];
      v8bf lo = *(const v8bf*)(p);
      v8bf hi = *(const v8bf*)(p + 16);
      qa[mi][ki] = __builtin_shufflevector(lo, hi, 0,1,2,3,4,5,6,7,8,9,10,11,12,13,14,15);
    }

  const float scale = 0.125f;  // 1/sqrt(64)
  const int rb = (lane >> 4) << 3, cl = lane & 15;
#pragma unroll 1
  for (int ns = 0; ns < 8; ++ns) {
    int key0 = w * 128 + ns * 16;
    v16bf b0 = load_b32x16(Kc, 64, 0,  key0, lane);  // contraction over head-dim
    v16bf b1 = load_b32x16(Kc, 64, 32, key0, lane);
#pragma unroll
    for (int mi = 0; mi < 4; ++mi) {
      v8f c = {};
      c = WMMA_BF16(qa[mi][0], b0, c);
      c = WMMA_BF16(qa[mi][1], b1, c);
      int row = mi * 16 + rb;
#pragma unroll
      for (int i = 0; i < 8; ++i)
        P[(row + i) * 1024 + key0 + cl] = c[i] * scale;
    }
  }
  __syncthreads();

  // ---- Phase 2: row softmax (64 rows x 1024). thread -> (row = t/4, quarter = t%4)
  const int srow = threadIdx.x >> 2, qtr = threadIdx.x & 3;
  float* pr = &P[srow * 1024 + qtr * 256];
  float lm = -1e30f;
  for (int j = 0; j < 256; j += 4) {
    v4f v = *(const v4f*)(pr + j);
    lm = fmaxf(lm, fmaxf(fmaxf(v.x, v.y), fmaxf(v.z, v.w)));
  }
  red[qtr][srow] = lm;
  __syncthreads();
  if (threadIdx.x < 64)
    rowstat[threadIdx.x] = fmaxf(fmaxf(red[0][threadIdx.x], red[1][threadIdx.x]),
                                 fmaxf(red[2][threadIdx.x], red[3][threadIdx.x]));
  __syncthreads();
  float rm = rowstat[srow];
  float ls = 0.0f;
  for (int j = 0; j < 256; ++j) {
    float e = __expf(pr[j] - rm);
    pr[j] = e;
    ls += e;
  }
  __syncthreads();  // rowstat reuse hazard
  red[qtr][srow] = ls;
  __syncthreads();
  if (threadIdx.x < 64)
    rowstat[threadIdx.x] = red[0][threadIdx.x] + red[1][threadIdx.x] +
                           red[2][threadIdx.x] + red[3][threadIdx.x];
  __syncthreads();
  float inv = 1.0f / rowstat[srow];
  float* gout = attn_out + ((size_t)bh * 1024 + qblk * 64 + srow) * 1024 + qtr * 256;
  for (int j = 0; j < 256; j += 4) {
    v4f v = *(const v4f*)(pr + j);
    v *= inv;
    *(v4f*)(pr + j) = v;       // keep normalized P in LDS for phase 3
    *(v4f*)(gout + j) = v;     // the one mandatory attention store
  }
  __syncthreads();

  // ---- Phase 3: context = P @ V (K dim = 1024).
  // Wave w: m tile = w&3; n tiles = (w>>2) and (w>>2)+2 -> one A build feeds 2 WMMAs.
  const int b = bh >> 4, h = bh & 15;
  const int mi  = w & 3;
  const int ni0 = (w >> 2) * 16;        // 0 or 16
  const int ni1 = ni0 + 32;             // 32 or 48
  v8f c0 = {}, c1 = {};
#pragma unroll 1
  for (int k0 = 0; k0 < 1024; k0 += 32) {
    // A fragment from LDS probabilities, converted f32 -> bf16 (shared by both tiles)
    int r = mi * 16 + (lane & 15);
    const float* pa = &P[r * 1024 + k0 + rb];
    v16bf a;
#pragma unroll
    for (int e = 0; e < 8; ++e) a[e] = (bf16_t)pa[e];
#pragma unroll
    for (int e = 0; e < 8; ++e) a[8 + e] = (bf16_t)pa[16 + e];
    v16bf bf0 = load_b32x16(Vb, 1024, k0, ni0, lane);  // Vt rows contiguous over keys
    v16bf bf1 = load_b32x16(Vb, 1024, k0, ni1, lane);
    c0 = WMMA_BF16(a, bf0, c0);
    c1 = WMMA_BF16(a, bf1, c1);
  }
#pragma unroll
  for (int i = 0; i < 8; ++i) {
    int rr = qblk * 64 + mi * 16 + rb + i;
    size_t rowoff = ((size_t)b * 1024 + rr) * 1024 + h * 64;
    vals[rowoff + ni0 + cl] = (bf16_t)c0[i];
    vals[rowoff + ni1 + cl] = (bf16_t)c1[i];
  }
}

// ---------------------------------------------------------------------------
// Output projection: out = vals @ W_o + b_o   (f32 out, straight into d_out)
// ---------------------------------------------------------------------------
__global__ void __launch_bounds__(256) o_gemm_kernel(const bf16_t* __restrict__ vals,
                                                     const bf16_t* __restrict__ wot,
                                                     const float* __restrict__ bias,
                                                     float* __restrict__ out) {
  const int lane = threadIdx.x & 31;
  const int w    = threadIdx.x >> 5;
  const int tm = blockIdx.x * 128, tn = blockIdx.y * 128;
  const int wm = (w & 1) * 64, wn = (w >> 1) * 32;

  v8f acc[4][2] = {};
#pragma unroll 1
  for (int k0 = 0; k0 < 1024; k0 += 32) {
    __builtin_prefetch(vals + (size_t)(tm + wm) * 1024 + k0 + 64, 0, 1);
    __builtin_prefetch(wot + (size_t)(tn + wn) * 1024 + k0 + 64, 0, 1);
    v16bf a[4];
#pragma unroll
    for (int mi = 0; mi < 4; ++mi)
      a[mi] = load_a16x32(vals, 1024, tm + wm + mi * 16, k0, lane);
    v16bf b0 = load_b32x16(wot, 1024, k0, tn + wn + 0, lane);
    v16bf b1 = load_b32x16(wot, 1024, k0, tn + wn + 16, lane);
#pragma unroll
    for (int mi = 0; mi < 4; ++mi) {
      acc[mi][0] = WMMA_BF16(a[mi], b0, acc[mi][0]);
      acc[mi][1] = WMMA_BF16(a[mi], b1, acc[mi][1]);
    }
  }

  const int rb = (lane >> 4) << 3, cl = lane & 15;
#pragma unroll
  for (int mi = 0; mi < 4; ++mi)
#pragma unroll
    for (int nj = 0; nj < 2; ++nj)
#pragma unroll
      for (int i = 0; i < 8; ++i) {
        int m = tm + wm + mi * 16 + rb + i;
        int n = tn + wn + nj * 16 + cl;
        out[(size_t)m * 1024 + n] = acc[mi][nj][i] + bias[n];
      }
}

// ---------------------------------------------------------------------------
extern "C" void kernel_launch(void* const* d_in, const int* in_sizes, int n_in,
                              void* d_out, int out_size, void* d_ws, size_t ws_size,
                              hipStream_t stream) {
  (void)in_sizes; (void)n_in; (void)out_size; (void)ws_size;
  const float* x     = (const float*)d_in[0];
  const float* W_qkv = (const float*)d_in[1];
  const float* b_qkv = (const float*)d_in[2];
  const float* W_o   = (const float*)d_in[3];
  const float* b_o   = (const float*)d_in[4];

  float* out  = (float*)d_out;                       // [8,1024,1024]
  float* attn = out + (size_t)8 * 1024 * 1024;       // [8,16,1024,1024]

  const size_t MD  = (size_t)8 * 1024 * 1024;        // 8388608
  char* ws = (char*)d_ws;
  bf16_t* xbf   = (bf16_t*)ws;  ws += MD * 2;
  bf16_t* wqkvt = (bf16_t*)ws;  ws += (size_t)3072 * 1024 * 2;
  bf16_t* wot   = (bf16_t*)ws;  ws += (size_t)1024 * 1024 * 2;
  bf16_t* Qb    = (bf16_t*)ws;  ws += MD * 2;
  bf16_t* Kb    = (bf16_t*)ws;  ws += MD * 2;
  bf16_t* Vt    = (bf16_t*)ws;  ws += MD * 2;
  bf16_t* vals  = (bf16_t*)ws;  ws += MD * 2;

  cvt_bf16_kernel<<<2048, 256, 0, stream>>>(x, xbf, (int)MD);
  cvt_t_bf16_kernel<<<1024, 256, 0, stream>>>(W_qkv, wqkvt, 1024, 3072);
  cvt_t_bf16_kernel<<<512, 256, 0, stream>>>(W_o, wot, 1024, 1024);

  qkv_gemm_kernel<<<dim3(64, 24), 256, 0, stream>>>(xbf, wqkvt, b_qkv, Qb, Kb, Vt);

  attn_kernel<<<dim3(128, 16), 256, 64 * 1024 * sizeof(float), stream>>>(
      Qb, Kb, Vt, attn, vals);

  o_gemm_kernel<<<dim3(64, 8), 256, 0, stream>>>(vals, wot, b_o, out);
}
